// GATEncoder_20641612824879
// MI455X (gfx1250) — matrix operand order repaired and checked by reading, not compiled, not run on hardware
//
#include <hip/hip_runtime.h>
#include <hip/hip_bf16.h>
#include <math.h>

// GATEncoder for MI455X (gfx1250, wave32, WMMA).
// Dense GEMMs: bf16 WMMA (v_wmma_f32_16x16x32_bf16), fp32 accumulate.
//   B panel staged in LDS, swizzled so each lane's fragment = 2x ds_load_b128
//   straight into the WMMA operand registers. 4 N-tiles per wave, K unrolled.
// Edge softmax + aggregation: fp32, atomic-based; scatter target fits in 192MB L2.

#define HEADS 2
#define HID   128
#define FOUT  256   // HEADS*HID

typedef __attribute__((ext_vector_type(16))) __bf16 v16bf;
typedef __attribute__((ext_vector_type(8)))  float  v8f;

// ---- order-preserving float <-> uint encoding for atomicMax on floats ----
__device__ __forceinline__ unsigned fenc(float f) {
  unsigned u = __float_as_uint(f);
  return (u & 0x80000000u) ? ~u : (u | 0x80000000u);
}
__device__ __forceinline__ float fdec(unsigned k) {
  return (k & 0x80000000u) ? __uint_as_float(k & 0x7FFFFFFFu) : __uint_as_float(~k);
}
#define ENC_NEG_INF 0x007FFFFFu   // fenc(-inf)

// =====================================================================
// GEMM: C[M,Nc] = A[M,K] @ B[K,Nc] (+ bias).
// Block = 256 threads (8 waves). Wave w -> 16 rows, block -> 64 columns
// (4 WMMA tiles per wave).
//
// LDS B layout (packed bf16 K-pairs, one unsigned per pair):
//   Bs[ ((kb*2 + hf)*2 + q)*256 + col*4 + vv ]
//     kb = k0/32, hf = lane half, q = fragment half (v/4), vv = v%4,
//     pair value = { bf16 B[K=k0+16*hf+2*v][col], bf16 B[K+1][col] }.
// A lane's fragment = two contiguous uint4 -> two ds_load_b128, no shuffling.
// Requires: M % 16 == 0 within grid coverage, Nc % 64 == 0, K % 32 == 0.
// =====================================================================
template <int K>
__global__ __launch_bounds__(256)
void gemm_wmma_bf16(const float* __restrict__ A, const float* __restrict__ B,
                    const float* __restrict__ bias, float* __restrict__ C,
                    int M, int Nc) {
  __shared__ unsigned Bs[K * 32];   // (K/32) * 2 * 2 * 64 * 4 dwords

  const int t = threadIdx.x;
  const int colBase = blockIdx.x * 64;

  // ---- cooperative staging of the B panel (coalesced on columns) ----
  for (int idx = t; idx < (K / 2) * 64; idx += 256) {
    int kp = idx >> 6;                  // K-pair index: K = 2*kp, 2*kp+1
    int c  = idx & 63;
    float b0 = B[(size_t)(2 * kp)     * Nc + colBase + c];
    float b1 = B[(size_t)(2 * kp + 1) * Nc + colBase + c];
    union { __bf16 h[2]; unsigned u; } pk;
    pk.h[0] = (__bf16)b0;
    pk.h[1] = (__bf16)b1;
    // decompose kp = kb*16 + hf*8 + v
    int kb = kp >> 4;
    int hf = (kp >> 3) & 1;
    int v  = kp & 7;
    Bs[(((kb * 2 + hf) * 2 + (v >> 2)) * 64 + c) * 4 + (v & 3)] = pk.u;
  }
  __syncthreads();

  const int lane = t & 31;
  const int wid  = t >> 5;
  const int l16  = lane & 15;
  const int hf   = lane >> 4;                 // lane half: 0 or 1
  const int tileM = blockIdx.y * 8 + wid;
  if (tileM * 16 >= M) return;                // wave-uniform; after the barrier

  const int rowA = tileM * 16 + l16;          // A row held by this lane
  const float2* arow = reinterpret_cast<const float2*>(A + (size_t)rowA * K);

  v8f acc0 = {}, acc1 = {}, acc2 = {}, acc3 = {};
#pragma unroll
  for (int k0 = 0; k0 < K; k0 += 32) {
    const int kb = k0 >> 5;
    // A fragment, contiguous k order (same as the B panel):
    //   VGPR v -> K = 16*hf + 2v
    union { v16bf v; __bf16 h[16]; } a;
#pragma unroll
    for (int v = 0; v < 8; ++v) {
      int kk = k0 + 16 * hf + 2 * v;
      float2 p = arow[kk >> 1];
      a.h[2 * v]     = (__bf16)p.x;
      a.h[2 * v + 1] = (__bf16)p.y;
    }
    // 4 N-tiles share the A fragment; each B fragment = 2x ds_load_b128
    const unsigned* bbase = &Bs[((kb * 2 + hf) * 2) * 256];
#pragma unroll
    for (int j = 0; j < 4; ++j) {
      union { v16bf v; uint4 q[2]; } b;
      const int cofs = (j * 16 + l16) * 4;
      b.q[0] = *reinterpret_cast<const uint4*>(bbase + cofs);         // v = 0..3
      b.q[1] = *reinterpret_cast<const uint4*>(bbase + 256 + cofs);   // v = 4..7
      v8f& acc = (j == 0) ? acc0 : (j == 1) ? acc1 : (j == 2) ? acc2 : acc3;
      acc = __builtin_amdgcn_wmma_f32_16x16x32_bf16(
          /*neg_a=*/false, a.v, /*neg_b=*/false, b.v,
          /*c_mod=*/(short)0, acc, /*reuse_a=*/false, /*reuse_b=*/false);
    }
  }

  // Epilogue: D layout -> VGPR v holds row 8*hf + v, column l16 (per tile)
#pragma unroll
  for (int j = 0; j < 4; ++j) {
    const v8f& acc = (j == 0) ? acc0 : (j == 1) ? acc1 : (j == 2) ? acc2 : acc3;
    int col = colBase + j * 16 + l16;
    float bv = bias ? bias[col] : 0.0f;
#pragma unroll
    for (int v = 0; v < 8; ++v) {
      int r = tileM * 16 + 8 * hf + v;
      C[(size_t)r * Nc + col] = acc[v] + bv;
    }
  }
}

// =====================================================================
// Per-layer state init: aggr = 0, emax = enc(-inf), denom = 0
// =====================================================================
__global__ void init_layer(float* __restrict__ aggr, unsigned* __restrict__ emax,
                           float* __restrict__ denom, int N) {
  int tid = blockIdx.x * blockDim.x + threadIdx.x;
  if (tid < N * FOUT) aggr[tid] = 0.0f;
  if (tid < N * HEADS) { emax[tid] = ENC_NEG_INF; denom[tid] = 0.0f; }
}

// =====================================================================
// a_src[n,h] = dot(xl[n,h,:], att_src[h,:]); same for a_dst.
// One wave per (n,h); coalesced loads + shuffle reduction.
// =====================================================================
__global__ __launch_bounds__(256)
void attn_coef(const float* __restrict__ xl, const float* __restrict__ att_src,
               const float* __restrict__ att_dst, float* __restrict__ a_src,
               float* __restrict__ a_dst, int N) {
  int wv   = (blockIdx.x * blockDim.x + threadIdx.x) >> 5;
  int lane = threadIdx.x & 31;
  int n = wv >> 1, h = wv & 1;
  if (n >= N) return;
  const float* row = xl + (size_t)n * FOUT + h * HID;
  const float* as  = att_src + h * HID;
  const float* ad  = att_dst + h * HID;
  float ss = 0.0f, sd = 0.0f;
  for (int c = lane; c < HID; c += 32) {
    float v = row[c];
    ss += v * as[c];
    sd += v * ad[c];
  }
#pragma unroll
  for (int off = 16; off > 0; off >>= 1) {
    ss += __shfl_xor(ss, off, 32);
    sd += __shfl_xor(sd, off, 32);
  }
  if (lane == 0) { a_src[n * 2 + h] = ss; a_dst[n * 2 + h] = sd; }
}

// =====================================================================
// Pass 1: segment max of leaky_relu(a_src[s]+a_dst[d]) over dst.
// i in [0,E) -> real edges; i in [E, E+N) -> self loop (s=d=i-E).
// =====================================================================
__global__ void edge_max(const int* __restrict__ esrc, const int* __restrict__ edst,
                         const float* __restrict__ a_src, const float* __restrict__ a_dst,
                         unsigned* __restrict__ emax, int E, int N) {
  int tid = blockIdx.x * blockDim.x + threadIdx.x;
  int total = (E + N) * HEADS;
  if (tid >= total) return;
  int i = tid >> 1, h = tid & 1;
  int s, d;
  if (i < E) { s = esrc[i]; d = edst[i]; } else { s = d = i - E; }
  float e = a_src[s * 2 + h] + a_dst[d * 2 + h];
  e = (e > 0.0f) ? e : 0.2f * e;
  atomicMax(&emax[d * 2 + h], fenc(e));
}

// =====================================================================
// Pass 2: stage exp(e - max) per (edge,head), accumulate segment sum.
// =====================================================================
__global__ void edge_expsum(const int* __restrict__ esrc, const int* __restrict__ edst,
                            const float* __restrict__ a_src, const float* __restrict__ a_dst,
                            const unsigned* __restrict__ emax, float* __restrict__ denom,
                            float* __restrict__ estage, int E, int N) {
  int tid = blockIdx.x * blockDim.x + threadIdx.x;
  int total = (E + N) * HEADS;
  if (tid >= total) return;
  int i = tid >> 1, h = tid & 1;
  int s, d;
  if (i < E) { s = esrc[i]; d = edst[i]; } else { s = d = i - E; }
  float e = a_src[s * 2 + h] + a_dst[d * 2 + h];
  e = (e > 0.0f) ? e : 0.2f * e;
  float ex = __expf(e - fdec(emax[d * 2 + h]));
  estage[tid] = ex;
  atomicAdd(&denom[d * 2 + h], ex);
}

// =====================================================================
// Pass 3: aggr[d,h,:] += xl[s,h,:] * alpha.
// 64 threads per edge, float4 gathers (global_load_b128), 4 edges/block.
// Channels 4u..4u+3 stay inside one head (128/4 = 32 threads per head).
// =====================================================================
__global__ __launch_bounds__(256)
void edge_msg(const int* __restrict__ esrc, const int* __restrict__ edst,
              const float* __restrict__ xl, const float* __restrict__ estage,
              const float* __restrict__ denom, float* __restrict__ aggr,
              int E, int ET) {
  int g = threadIdx.x >> 6;                   // edge slot within block
  int i = blockIdx.x * 4 + g;
  if (i >= ET) return;
  int u = threadIdx.x & 63;                   // channel group: 4u..4u+3
  int h = u >> 5;
  int s, d;
  if (i < E) { s = esrc[i]; d = edst[i]; } else { s = d = i - E; }
  float alpha = estage[i * 2 + h] / (denom[d * 2 + h] + 1e-16f);
  const float4 xv = *reinterpret_cast<const float4*>(xl + (size_t)s * FOUT + u * 4);
  float* ap = aggr + (size_t)d * FOUT + u * 4;
  atomicAdd(ap + 0, xv.x * alpha);
  atomicAdd(ap + 1, xv.y * alpha);
  atomicAdd(ap + 2, xv.z * alpha);
  atomicAdd(ap + 3, xv.w * alpha);
}

// =====================================================================
// Head mean + bias + exact GELU -> h[N,HID]
// =====================================================================
__global__ void finalize_gelu(const float* __restrict__ aggr, const float* __restrict__ bias,
                              float* __restrict__ hout, int N) {
  int tid = blockIdx.x * blockDim.x + threadIdx.x;
  if (tid >= N * HID) return;
  int n = tid >> 7, c = tid & 127;
  float m = 0.5f * (aggr[(size_t)n * FOUT + c] + aggr[(size_t)n * FOUT + HID + c]) + bias[c];
  hout[tid] = 0.5f * m * (1.0f + erff(m * 0.7071067811865476f));
}

// =====================================================================
// Host-side orchestration
// =====================================================================
extern "C" void kernel_launch(void* const* d_in, const int* in_sizes, int n_in,
                              void* d_out, int out_size, void* d_ws, size_t ws_size,
                              hipStream_t stream) {
  const float* x        = (const float*)d_in[0];
  const int*   eidx     = (const int*)d_in[1];     // [2,E] flat: src then dst
  const float* W0       = (const float*)d_in[2];
  const float* att_src0 = (const float*)d_in[3];
  const float* att_dst0 = (const float*)d_in[4];
  const float* bias0    = (const float*)d_in[5];
  const float* W1       = (const float*)d_in[6];
  const float* att_src1 = (const float*)d_in[7];
  const float* att_dst1 = (const float*)d_in[8];
  const float* bias1    = (const float*)d_in[9];
  const float* proj_w   = (const float*)d_in[10];
  const float* proj_b   = (const float*)d_in[11];
  float* out = (float*)d_out;

  const int FIN = 256;
  const int N = in_sizes[0] / FIN;
  const int E = in_sizes[1] / 2;
  const int ET = E + N;                 // edges incl. self loops
  const int* esrc = eidx;
  const int* edst = eidx + E;

  // ---- carve workspace ----
  char* base = (char*)d_ws;
  size_t off = 0;
  auto alloc = [&](size_t bytes) -> void* {
    void* p = base + off;
    off += (bytes + 255) & ~(size_t)255;
    return p;
  };
  float*    xl     = (float*)alloc((size_t)N * FOUT * sizeof(float));
  float*    aggr   = (float*)alloc((size_t)N * FOUT * sizeof(float));
  float*    hbuf   = (float*)alloc((size_t)N * HID * sizeof(float));
  float*    a_src  = (float*)alloc((size_t)N * HEADS * sizeof(float));
  float*    a_dst  = (float*)alloc((size_t)N * HEADS * sizeof(float));
  unsigned* emax   = (unsigned*)alloc((size_t)N * HEADS * sizeof(unsigned));
  float*    denom  = (float*)alloc((size_t)N * HEADS * sizeof(float));
  float*    estage = (float*)alloc((size_t)ET * HEADS * sizeof(float));
  (void)ws_size; (void)n_in; (void)out_size;

  const int mtiles = (N + 15) / 16;
  const dim3 gemmGridL(FOUT / 64, (mtiles + 7) / 8);   // layers: Nc = 256
  const dim3 gemmGridP(HID / 64, (mtiles + 7) / 8);    // projection: Nc = 128
  const int initBlocks = (N * FOUT + 255) / 256;
  const int attnBlocks = (N * HEADS * 32 + 255) / 256;
  const int edgeBlocks = (ET * HEADS + 255) / 256;
  const int msgBlocks  = (ET + 3) / 4;
  const int finBlocks  = (N * HID + 255) / 256;

  auto run_attention = [&](const float* as, const float* ad, const float* bi) {
    init_layer<<<initBlocks, 256, 0, stream>>>(aggr, emax, denom, N);
    attn_coef<<<attnBlocks, 256, 0, stream>>>(xl, as, ad, a_src, a_dst, N);
    edge_max<<<edgeBlocks, 256, 0, stream>>>(esrc, edst, a_src, a_dst, emax, E, N);
    edge_expsum<<<edgeBlocks, 256, 0, stream>>>(esrc, edst, a_src, a_dst, emax, denom, estage, E, N);
    edge_msg<<<msgBlocks, 256, 0, stream>>>(esrc, edst, xl, estage, denom, aggr, E, ET);
    finalize_gelu<<<finBlocks, 256, 0, stream>>>(aggr, bi, hbuf, N);
  };

  // Layer 0: x[N,256] -> hbuf[N,128]
  gemm_wmma_bf16<256><<<gemmGridL, 256, 0, stream>>>(x, W0, nullptr, xl, N, FOUT);
  run_attention(att_src0, att_dst0, bias0);
  // Layer 1: hbuf[N,128] -> hbuf[N,128] (gemm consumes hbuf before finalize rewrites it)
  gemm_wmma_bf16<128><<<gemmGridL, 256, 0, stream>>>(hbuf, W1, nullptr, xl, N, FOUT);
  run_attention(att_src1, att_dst1, bias1);
  // Projection: out = hbuf @ proj_w + proj_b
  gemm_wmma_bf16<128><<<gemmGridP, 256, 0, stream>>>(hbuf, proj_w, proj_b, out, N, HID);
}
